// MotionEstimator_22162031247703
// MI455X (gfx1250) — matrix operand order repaired and checked by reading, not compiled, not run on hardware
//
#include <hip/hip_runtime.h>
#include <hip/hip_bf16.h>

typedef __bf16 bhalf;
typedef __attribute__((ext_vector_type(16))) __bf16 v16bf;
typedef __attribute__((ext_vector_type(8)))  __bf16 v8bf;
typedef __attribute__((ext_vector_type(8)))  float  v8f;

typedef unsigned int u32x4 __attribute__((ext_vector_type(4)));
typedef int          i32x4 __attribute__((ext_vector_type(4)));
typedef int          i32x8 __attribute__((ext_vector_type(8)));

#if defined(__has_builtin)
#if __has_builtin(__builtin_amdgcn_tensor_load_to_lds) && \
    __has_builtin(__builtin_amdgcn_s_wait_tensorcnt)
#define USE_TDM 1
#endif
#endif
#ifndef USE_TDM
#define USE_TDM 0
#endif

#define BB 4
#define HH 192
#define WW 320
#define HWSZ (HH * WW)          // 61440

// ---------------------------------------------------------------------------
// 1) softsplat scatter: atomically splat 64 feature channels + a ones channel
//    to the 4 bilinear corners of (x+u, y+v), u,v = 0.125 * last_flow[ch..ch+1]
// ---------------------------------------------------------------------------
__global__ __launch_bounds__(256)
void splat_scatter_kernel(const float* __restrict__ feat,
                          const float* __restrict__ flow, int flowChBase,
                          float* __restrict__ accum)   // [B][65][H][W]
{
    int gid = blockIdx.x * blockDim.x + threadIdx.x;
    if (gid >= BB * HWSZ) return;
    int b = gid / HWSZ;
    int p = gid - b * HWSZ;
    int y = p / WW;
    int x = p - y * WW;

    float u = flow[((size_t)b * 4 + flowChBase) * HWSZ + p] * 0.125f;
    float v = flow[((size_t)b * 4 + flowChBase + 1) * HWSZ + p] * 0.125f;
    float tx = (float)x + u, ty = (float)y + v;
    float fx0 = floorf(tx), fy0 = floorf(ty);
    int ix0 = (int)fx0, iy0 = (int)fy0;
    float fx = tx - fx0, fy = ty - fy0;

    float wgt[4] = { (1.f - fx) * (1.f - fy), fx * (1.f - fy),
                     (1.f - fx) * fy,         fx * fy };
    int   xs[4]  = { ix0, ix0 + 1, ix0,     ix0 + 1 };
    int   ys[4]  = { iy0, iy0,     iy0 + 1, iy0 + 1 };
    bool  ok[4];
    int   idx[4];
    #pragma unroll
    for (int k = 0; k < 4; ++k) {
        ok[k]  = (xs[k] >= 0) && (xs[k] < WW) && (ys[k] >= 0) && (ys[k] < HH);
        idx[k] = ok[k] ? (ys[k] * WW + xs[k]) : 0;
    }

    const float* fsrc  = feat  + (size_t)b * 64 * HWSZ + p;
    float*       abase = accum + (size_t)b * 65 * HWSZ;

    for (int c = 0; c < 64; ++c) {
        float fv = fsrc[(size_t)c * HWSZ];
        #pragma unroll
        for (int k = 0; k < 4; ++k)
            if (ok[k]) atomicAdd(abase + (size_t)c * HWSZ + idx[k], fv * wgt[k]);
    }
    #pragma unroll
    for (int k = 0; k < 4; ++k)
        if (ok[k]) atomicAdd(abase + (size_t)64 * HWSZ + idx[k], wgt[k]);
}

// ---------------------------------------------------------------------------
// 2) normalize splat: out = accum[c]/max(accum[64],!=0); write f32 (for corr)
//    and bf16 into the concat buffer x [B][288][H][W] at channel offset xoff
// ---------------------------------------------------------------------------
__global__ __launch_bounds__(256)
void normalize_splat_kernel(const float* __restrict__ accum,
                            float* __restrict__ fdst,
                            bhalf* __restrict__ xbuf, int xoff)
{
    int gid = blockIdx.x * blockDim.x + threadIdx.x;
    if (gid >= BB * 64 * HWSZ) return;
    int b = gid / (64 * HWSZ);
    int r = gid - b * 64 * HWSZ;
    int c = r / HWSZ;
    int p = r - c * HWSZ;

    float norm = accum[((size_t)b * 65 + 64) * HWSZ + p];
    norm = (norm == 0.f) ? 1.f : norm;
    float v = accum[((size_t)b * 65 + c) * HWSZ + p] / norm;
    fdst[((size_t)b * 64 + c) * HWSZ + p] = v;
    xbuf[((size_t)b * 288 + xoff + c) * HWSZ + p] = (bhalf)v;
}

// ---------------------------------------------------------------------------
// 3) fill concat tail: channels 209..272 = last_feat, 273..276 = last_flow,
//    277..287 = zero pad (K padding for the 1x1 WMMA GEMM)
// ---------------------------------------------------------------------------
__global__ __launch_bounds__(256)
void fill_tail_kernel(const float* __restrict__ last_feat,
                      const float* __restrict__ last_flow,
                      bhalf* __restrict__ xbuf)
{
    int gid = blockIdx.x * blockDim.x + threadIdx.x;
    if (gid >= BB * 79 * HWSZ) return;
    int b  = gid / (79 * HWSZ);
    int r  = gid - b * 79 * HWSZ;
    int cc = r / HWSZ;
    int p  = r - cc * HWSZ;
    int ch = 209 + cc;

    float v = 0.f;
    if (ch < 273)      v = last_feat[((size_t)b * 64 + (ch - 209)) * HWSZ + p];
    else if (ch < 277) v = last_flow[((size_t)b * 4  + (ch - 273)) * HWSZ + p];
    xbuf[((size_t)b * 288 + ch) * HWSZ + p] = (bhalf)v;
}

// ---------------------------------------------------------------------------
// 4) cost volume: 81 displacements in [-4,4]^2, mean over 64 ch, lrelu(0.1),
//    bf16 into x channels 0..80. f0 cached in registers; f1 shifted reads hit L2.
// ---------------------------------------------------------------------------
__global__ __launch_bounds__(128)
void corr_kernel(const float* __restrict__ f0, const float* __restrict__ f1,
                 bhalf* __restrict__ xbuf)
{
    int gid = blockIdx.x * blockDim.x + threadIdx.x;
    if (gid >= BB * HWSZ) return;
    int b = gid / HWSZ;
    int p = gid - b * HWSZ;
    int y = p / WW;
    int x = p - y * WW;

    float r0[64];
    const float* f0p = f0 + (size_t)b * 64 * HWSZ + p;
    #pragma unroll
    for (int c = 0; c < 64; ++c) r0[c] = f0p[(size_t)c * HWSZ];

    const float* f1b = f1 + (size_t)b * 64 * HWSZ;
    bhalf* xo = xbuf + (size_t)b * 288 * HWSZ + p;

    int d = 0;
    for (int dy = -4; dy <= 4; ++dy) {
        for (int dx = -4; dx <= 4; ++dx, ++d) {
            int yy = y + dy, xx = x + dx;
            float s = 0.f;
            if ((unsigned)yy < (unsigned)HH && (unsigned)xx < (unsigned)WW) {
                const float* f1p = f1b + (size_t)yy * WW + xx;
                #pragma unroll
                for (int c = 0; c < 64; ++c) s += r0[c] * f1p[(size_t)c * HWSZ];
            }
            s *= (1.f / 64.f);
            s = (s > 0.f) ? s : 0.1f * s;       // leaky_relu on the cost volume
            xo[(size_t)d * HWSZ] = (bhalf)s;
        }
    }
}

// ---------------------------------------------------------------------------
// 5) pack OIHW f32 weights -> bf16 [CoutPad][KtotPad], tap-major K = t*Cp + c,
//    zero padded in Cout (mult of 64), per-tap channels (to Cp) and K (to 64)
// ---------------------------------------------------------------------------
__global__ __launch_bounds__(256)
void pack_w_kernel(const float* __restrict__ w, bhalf* __restrict__ wp,
                   int Cout, int CoutPad, int Cin, int Cp, int taps,
                   int Ktot, int KtotPad)
{
    int gid = blockIdx.x * blockDim.x + threadIdx.x;
    if (gid >= CoutPad * KtotPad) return;
    int o = gid / KtotPad;
    int k = gid - o * KtotPad;
    float v = 0.f;
    if (o < Cout && k < Ktot) {
        int t = k / Cp;
        int c = k - t * Cp;
        if (c < Cin)
            v = (taps == 1) ? w[(size_t)o * Cin + c]
                            : w[((size_t)o * Cin + c) * 9 + t];
    }
    wp[gid] = (bhalf)v;
}

// ---------------------------------------------------------------------------
// 6) implicit-GEMM conv via v_wmma_f32_16x16x32_bf16.
//    Block = 128 threads (4 waves) = 32 pixels x 64 couts. Each wave:
//    1 cout tile x 2 pixel tiles, K-chunks processed in pairs -> 4 WMMAs per
//    barrier round. B-operand staging:
//      * interior chunks: Tensor Data Mover (tensor_load_to_lds, one wave per
//        chunk issues a 32x32 2-byte-element D# with row stride H*W), waited
//        on TENSORcnt
//      * border chunks / no-TDM toolchain: cooperative 16B global loads ->
//        contiguous 16B ds_store (LDS laid out [k][pixel] = TDM tile order)
// ---------------------------------------------------------------------------
__global__ __launch_bounds__(128)
void conv_wmma_kernel(const bhalf* __restrict__ in,      // [B][Cin][H][W]
                      const bhalf* __restrict__ wts,     // [CoutPad][KtotPad]
                      const float* __restrict__ bias,
                      bhalf* __restrict__ outb,          // may be null
                      float* __restrict__ outf,          // may be null
                      int Cin, int Cp, int taps, int Ktot, int KtotPad,
                      int Cout, int relu)
{
    __shared__ bhalf lb[2][32][32];                      // [chunk][k][pixel] 4KB

    const int tid  = threadIdx.x;
    const int lane = tid & 31;
    const int wave = tid >> 5;
    const int n    = lane & 15;
    const int half = lane >> 4;

    // pixel tile: 32 consecutive pixels along W
    const int tilesPerRow = WW >> 5;                     // 10
    int pt   = blockIdx.x;
    int x0   = (pt % tilesPerRow) << 5;
    int rowi = pt / tilesPerRow;                         // b*H + y
    int y    = rowi % HH;
    int b    = rowi / HH;

    int coutTile = blockIdx.y * 4 + wave;
    const bhalf* wrow = wts + (size_t)(coutTile * 16 + n) * KtotPad;
    const size_t inB  = (size_t)b * Cin * HWSZ;

    v8f acc0 = {}, acc1 = {};
    const int npairs = KtotPad >> 6;

    // cooperative-staging thread mapping: kk = channel-in-chunk, 8 pixels
    const int kkL = tid >> 2;                            // 0..31
    const int xgL = (tid & 3) << 3;                      // 0,8,16,24

    // TDM is usable when the dx-shifted 32-pixel window is interior in x for
    // any dx in {-1,0,1}; x0 is a multiple of 32
    const bool interiorX = (x0 >= 32) && (x0 + 64 <= WW);

    int t = 0, c0 = 0;                                   // tap / channel base
    for (int pr = 0; pr < npairs; ++pr) {
        // ---- load phase: stage up to 2 chunks of B into LDS ----
        #pragma unroll
        for (int u = 0; u < 2; ++u) {
            int kbase = (pr << 6) + (u << 5);
            if (kbase < Ktot) {
                int dy = (taps == 9) ? (t / 3 - 1) : 0;
                int dx = (taps == 9) ? (t - (t / 3) * 3 - 1) : 0;
                int yy = y + dy;
                bool yok = (unsigned)yy < (unsigned)HH;
#if USE_TDM
                bool tdmok = interiorX && yok && (c0 + 32 <= Cin);
#else
                bool tdmok = false;
#endif
                if (tdmok) {
#if USE_TDM
                    if (wave == u) {
                        // D# group0: count=1, lds_addr, 57-bit global addr, type=2
                        const bhalf* gsrc =
                            in + inB + ((size_t)c0 * HH + yy) * WW + (x0 + dx);
                        unsigned long long ga = (unsigned long long)gsrc;
                        unsigned la = (unsigned)(size_t)&lb[u][0][0];
                        u32x4 g0 = { 1u, la, (unsigned)ga,
                                     (unsigned)(((ga >> 32) & 0x1FFFFFFu) | (2u << 30)) };
                        // D# group1: data_size=2B; tensor 32x32 tile 32x32,
                        // dim0 stride = H*W elements (channel stride)
                        i32x8 g1 = { 0x10000,
                                     (int)(32u << 16),   // tensor_dim0 = 32
                                     (int)(32u << 16),   // tensor_dim1 = 32
                                     (int)(32u << 16),   // tile_dim0  = 32
                                     32,                 // tile_dim1  = 32
                                     HWSZ,               // tensor_dim0_stride
                                     0, 0 };
                        i32x4 z4 = { 0, 0, 0, 0 };
#if __clang_major__ >= 23
                        i32x8 z8 = { 0, 0, 0, 0, 0, 0, 0, 0 };
                        __builtin_amdgcn_tensor_load_to_lds(g0, g1, z4, z4, z8, 0);
#else
                        __builtin_amdgcn_tensor_load_to_lds(g0, g1, z4, z4, 0);
#endif
                    }
#endif
                } else {
                    int c     = c0 + kkL;
                    bool rowok = (c < Cin) && yok;
                    int xbase  = x0 + xgL + dx;
                    const bhalf* src =
                        in + inB + ((size_t)c * HH + yy) * WW + xbase;
                    bhalf tmp[8];
                    if (rowok && dx == 0) {
                        *(v8bf*)tmp = *(const v8bf*)src;     // aligned 16B load
                    } else {
                        #pragma unroll
                        for (int j = 0; j < 8; ++j) {
                            int xx = xbase + j;
                            tmp[j] = (rowok && (unsigned)xx < (unsigned)WW)
                                         ? src[j] : (bhalf)0.f;
                        }
                    }
                    *(v8bf*)&lb[u][kkL][xgL] = *(v8bf*)tmp;  // 16B ds_store
                }
                // advance tap/channel base for next chunk
                c0 += 32;
                if (c0 >= Cp) { c0 = 0; ++t; }
            }
        }
#if USE_TDM
        __builtin_amdgcn_s_wait_tensorcnt(0);
#endif
        __syncthreads();

        // ---- compute phase: 2 chunks x (1 A frag, 2 B frags, 2 WMMA) ----
        #pragma unroll
        for (int u = 0; u < 2; ++u) {
            int kbase = (pr << 6) + (u << 5);
            if (kbase < Ktot) {
                __builtin_prefetch(wrow + kbase + 64, 0, 3);

                // A: 16-bit layout, M = lane%16; half0 K={0..7,16..23}, half1 +8
                v16bf afrag;
                const bhalf* wr = wrow + kbase + half * 8;
                #pragma unroll
                for (int e = 0; e < 8; ++e) {
                    afrag[e]     = wr[e];
                    afrag[8 + e] = wr[16 + e];
                }

                // B: N = lane%16; half0 K={0..7,16..23}, half1 +8, from [k][pixel] LDS tile
                v16bf b0, b1;
                #pragma unroll
                for (int e = 0; e < 16; ++e) {
                    b0[e] = lb[u][half * 8 + (e & 7) + ((e >> 3) << 4)][n];
                    b1[e] = lb[u][half * 8 + (e & 7) + ((e >> 3) << 4)][16 + n];
                }

                acc0 = __builtin_amdgcn_wmma_f32_16x16x32_bf16(
                           false, afrag, false, b0, (short)0, acc0, false, false);
                acc1 = __builtin_amdgcn_wmma_f32_16x16x32_bf16(
                           false, afrag, false, b1, (short)0, acc1, false, false);
            }
        }
        __syncthreads();
    }

    // C/D layout: VGPR r holds M = r + 8*half, N = lane%16
    #pragma unroll
    for (int r = 0; r < 8; ++r) {
        int co = coutTile * 16 + half * 8 + r;
        if (co < Cout) {
            float v0 = acc0[r] + bias[co];
            float v1 = acc1[r] + bias[co];
            if (relu) {
                v0 = (v0 > 0.f) ? v0 : 0.1f * v0;
                v1 = (v1 > 0.f) ? v1 : 0.1f * v1;
            }
            size_t o = (((size_t)b * Cout + co) * HH + y) * WW + x0 + n;
            if (outb) { outb[o] = (bhalf)v0; outb[o + 16] = (bhalf)v1; }
            if (outf) { outf[o] = v0;        outf[o + 16] = v1; }
        }
    }
}

// ---------------------------------------------------------------------------
// launcher
// ---------------------------------------------------------------------------
static inline size_t alignup(size_t v) { return (v + 255) & ~(size_t)255; }

extern "C" void kernel_launch(void* const* d_in, const int* in_sizes, int n_in,
                              void* d_out, int out_size, void* d_ws, size_t ws_size,
                              hipStream_t stream)
{
    const float* feat0     = (const float*)d_in[0];
    const float* feat1     = (const float*)d_in[1];
    const float* last_feat = (const float*)d_in[2];
    const float* last_flow = (const float*)d_in[3];
    const float* w[6]; const float* bset[6];
    for (int i = 0; i < 6; ++i) { w[i] = (const float*)d_in[4 + 2 * i]; bset[i] = (const float*)d_in[5 + 2 * i]; }
    float* out = (float*)d_out;
    float* out_flow = out;                               // B*4*H*W
    float* out_feat = out + (size_t)BB * 4 * HWSZ;       // B*64*H*W

    // workspace layout
    char* ws = (char*)d_ws;
    size_t off = 0;
    auto take = [&](size_t bytes) { size_t o = off; off = alignup(off + bytes); return o; };
    float* splat0 = (float*)(ws + take((size_t)BB * 65 * HWSZ * 4));
    float* splat1 = (float*)(ws + take((size_t)BB * 65 * HWSZ * 4));
    float* f0f    = (float*)(ws + take((size_t)BB * 64 * HWSZ * 4));
    float* f1f    = (float*)(ws + take((size_t)BB * 64 * HWSZ * 4));
    bhalf* xbuf   = (bhalf*)(ws + take((size_t)BB * 288 * HWSZ * 2));
    bhalf* act1   = (bhalf*)(ws + take((size_t)BB * 160 * HWSZ * 2));
    bhalf* act2   = (bhalf*)(ws + take((size_t)BB * 128 * HWSZ * 2));
    bhalf* act3   = (bhalf*)(ws + take((size_t)BB * 112 * HWSZ * 2));
    bhalf* act4   = (bhalf*)(ws + take((size_t)BB * 96  * HWSZ * 2));
    bhalf* act5   = (bhalf*)(ws + take((size_t)BB * 64  * HWSZ * 2));
    // per-layer GEMM geometry
    const int CO[6]  = { 160, 128, 112,  96, 64,  4 };   // Cout
    const int COP[6] = { 192, 128, 128, 128, 64, 64 };   // Cout padded (mult 64)
    const int CI[6]  = { 288, 160, 128, 112, 96, 64 };   // Cin (real, stored)
    const int CP[6]  = { 288, 160, 128, 128, 96, 64 };   // Cin padded per tap
    const int TP[6]  = {   1,   9,   9,   9,  9,  9 };   // taps
    int KT[6], KTP[6];
    bhalf* wp[6];
    for (int i = 0; i < 6; ++i) {
        KT[i]  = TP[i] * CP[i];
        KTP[i] = (KT[i] + 63) & ~63;                     // pairs of 32-chunks
        wp[i]  = (bhalf*)(ws + take((size_t)COP[i] * KTP[i] * 2));
    }

    // zero the atomic splat accumulators (contiguous pair)
    hipMemsetAsync(splat0, 0, (size_t)2 * BB * 65 * HWSZ * 4, stream);

    const int T = 256;
    int nPix = BB * HWSZ;

    // 1) scatter splats
    splat_scatter_kernel<<<(nPix + T - 1) / T, T, 0, stream>>>(feat0, last_flow, 0, splat0);
    splat_scatter_kernel<<<(nPix + T - 1) / T, T, 0, stream>>>(feat1, last_flow, 2, splat1);

    // 2) normalize -> f0/f1 f32 + bf16 into x (ch 81.. / 145..)
    int nFeat = BB * 64 * HWSZ;
    normalize_splat_kernel<<<(nFeat + T - 1) / T, T, 0, stream>>>(splat0, f0f, xbuf, 81);
    normalize_splat_kernel<<<(nFeat + T - 1) / T, T, 0, stream>>>(splat1, f1f, xbuf, 145);

    // 3) concat tail: last_feat, last_flow, zero K-pad
    int nTail = BB * 79 * HWSZ;
    fill_tail_kernel<<<(nTail + T - 1) / T, T, 0, stream>>>(last_feat, last_flow, xbuf);

    // 4) cost volume -> x channels 0..80
    corr_kernel<<<(nPix + 127) / 128, 128, 0, stream>>>(f0f, f1f, xbuf);

    // 5) pack all weights to bf16
    for (int i = 0; i < 6; ++i) {
        int nel = COP[i] * KTP[i];
        pack_w_kernel<<<(nel + T - 1) / T, T, 0, stream>>>(
            w[i], wp[i], CO[i], COP[i], CI[i], CP[i], TP[i], KT[i], KTP[i]);
    }

    // 6) conv chain on WMMA (32-pixel tiles)
    dim3 blk(128);
    int nTiles = BB * HH * (WW >> 5);                    // 7680 pixel tiles
    conv_wmma_kernel<<<dim3(nTiles, COP[0] / 64), blk, 0, stream>>>(xbuf, wp[0], bset[0], act1, nullptr, CI[0], CP[0], TP[0], KT[0], KTP[0], CO[0], 1);
    conv_wmma_kernel<<<dim3(nTiles, COP[1] / 64), blk, 0, stream>>>(act1, wp[1], bset[1], act2, nullptr, CI[1], CP[1], TP[1], KT[1], KTP[1], CO[1], 1);
    conv_wmma_kernel<<<dim3(nTiles, COP[2] / 64), blk, 0, stream>>>(act2, wp[2], bset[2], act3, nullptr, CI[2], CP[2], TP[2], KT[2], KTP[2], CO[2], 1);
    conv_wmma_kernel<<<dim3(nTiles, COP[3] / 64), blk, 0, stream>>>(act3, wp[3], bset[3], act4, nullptr, CI[3], CP[3], TP[3], KT[3], KTP[3], CO[3], 1);
    conv_wmma_kernel<<<dim3(nTiles, COP[4] / 64), blk, 0, stream>>>(act4, wp[4], bset[4], act5, out_feat, CI[4], CP[4], TP[4], KT[4], KTP[4], CO[4], 1);
    conv_wmma_kernel<<<dim3(nTiles, COP[5] / 64), blk, 0, stream>>>(act5, wp[5], bset[5], nullptr, out_flow, CI[5], CP[5], TP[5], KT[5], KTP[5], CO[5], 0);
}
